// MaskedDeformableAttention_55894704390251
// MI455X (gfx1250) — matrix operand order repaired and checked by reading, not compiled, not run on hardware
//
#include <hip/hip_runtime.h>
#include <hip/hip_bf16.h>
#include <cstdint>
#include <cstddef>

// ---------------------------------------------------------------------------
// Masked deformable attention for MI455X (gfx1250, wave32, WMMA)
// Pipeline: bf16 convert -> WMMA GEMMs (x@Wkv, q@Wq, q@Woff) with async
//           LDS-staged B tiles -> deformable sampling + softmax (VALU) ->
//           WMMA proj GEMM
// ---------------------------------------------------------------------------

typedef unsigned short ushort_t;
typedef __bf16 bf16_t;
typedef bf16_t v16bf __attribute__((ext_vector_type(16)));
typedef float  v8f   __attribute__((ext_vector_type(8)));
typedef int    v4i_t __attribute__((ext_vector_type(4)));

#define BB 4
#define NNN 576
#define TTT 3
#define CCC 768
#define HHH 12
#define PPP 9
#define DHH 64
#define HPD 24
#define WPD 24
#define SCALE_F 0.125f

// LDS B tile: 64 columns (n) x 64 deep (k) bf16, row stride padded to
// 72 ushorts (144 B) so fragment reads spread across banks. Double buffered.
#define BROW 72

#if defined(__has_builtin)
#if __has_builtin(__builtin_amdgcn_global_load_async_to_lds_b128) && \
    __has_builtin(__builtin_amdgcn_s_wait_asynccnt)
#define HAVE_ASYNC_LDS 1
#endif
#endif

// round-to-nearest-even f32 -> bf16
__device__ __forceinline__ ushort_t f2bf(float f) {
  union { float f; unsigned u; } c; c.f = f;
  unsigned r = c.u + 0x7FFFu + ((c.u >> 16) & 1u);
  return (ushort_t)(r >> 16);
}

__device__ __forceinline__ void async_copy_b128(const void* g, void* l) {
#ifdef HAVE_ASYNC_LDS
  typedef __attribute__((address_space(1))) v4i_t* gp_t;  // global int4*
  typedef __attribute__((address_space(3))) v4i_t* lp_t;  // LDS int4*
  __builtin_amdgcn_global_load_async_to_lds_b128((gp_t)g, (lp_t)l, 0, 0);
#else
  *(uint4*)l = *(const uint4*)g;
#endif
}

__device__ __forceinline__ void wait_async_done() {
#ifdef HAVE_ASYNC_LDS
  __builtin_amdgcn_s_wait_asynccnt(0);
#endif
}

// ---------------------------------------------------------------------------
// Conversion kernels
// ---------------------------------------------------------------------------
__global__ void cvt_bf16_kernel(const float* __restrict__ src,
                                ushort_t* __restrict__ dst, int n) {
  int i = blockIdx.x * blockDim.x + threadIdx.x;
  if (i < n) dst[i] = f2bf(src[i]);
}

// query = x[:, :, T/2, :]  ->  bf16 (rows = b*N+n, cols = C)
__global__ void extract_query_bf16(const float* __restrict__ x,
                                   ushort_t* __restrict__ dst, int n) {
  int i = blockIdx.x * blockDim.x + threadIdx.x;
  if (i >= n) return;
  int row = i / CCC, c = i % CCC;
  dst[i] = f2bf(x[((size_t)row * TTT + (TTT / 2)) * CCC + c]);
}

// W (K x Nsrc row-major f32) -> Wt (Npad x K row-major bf16), zero pad rows
__global__ void transpose_w_bf16(const float* __restrict__ W,
                                 ushort_t* __restrict__ Wt,
                                 int K, int Nsrc, int Npad) {
  int i = blockIdx.x * blockDim.x + threadIdx.x;
  if (i >= Npad * K) return;
  int nrow = i / K, k = i % K;
  float v = (nrow < Nsrc) ? W[(size_t)k * Nsrc + nrow] : 0.0f;
  Wt[i] = f2bf(v);
}

// ---------------------------------------------------------------------------
// WMMA bf16 GEMM:  C[M,N] = A[M,K] * Bt[N,K]^T  (+ bias[N])
// 256 threads = 8 waves; block tile 256 rows x 64 cols; each wave owns a
// 32x64 strip (2 M-tiles x 4 N-tiles). B is staged through a double-buffered
// LDS tile with async global->LDS copies (64 cols x 64 K per stage).
// grid = (N/64, M/256). K % 64 == 0, M % 256 == 0, N % 64 == 0.
// ---------------------------------------------------------------------------
union ABfrag { uint4 u[2]; v16bf v; };

__global__ __launch_bounds__(256) void gemm_bf16_wmma(
    const ushort_t* __restrict__ A, const ushort_t* __restrict__ Bt,
    float* __restrict__ C, const float* __restrict__ bias,
    int M, int N, int K) {
  __shared__ ushort_t ldsB[2][64 * BROW];

  const int tid  = threadIdx.x;
  const int lane = tid & 31;
  const int wave = tid >> 5;
  const int half = lane >> 4;   // 0: lanes 0-15, 1: lanes 16-31
  const int l16  = lane & 15;

  const int m0 = blockIdx.y * 256 + wave * 32;
  const int n0 = blockIdx.x * 64;

  v8f acc[2][4];
#pragma unroll
  for (int mt = 0; mt < 2; ++mt)
#pragma unroll
    for (int nt = 0; nt < 4; ++nt) acc[mt][nt] = (v8f){};

  const ushort_t* arow0 = A + (size_t)(m0 + l16) * K;
  const ushort_t* arow1 = arow0 + (size_t)16 * K;

  // staging assignment: 512 x 16B transfers per tile, 2 per thread
  const int srow0 = tid >> 3, sseg0 = tid & 7;           // q = tid
  const int srow1 = (tid + 256) >> 3, sseg1 = tid & 7;   // q = tid + 256

  const int ksteps = K >> 6;  // 64-deep K slices

  // stage slice 0 into buffer 0
  {
    const ushort_t* g0 = Bt + (size_t)(n0 + srow0) * K + sseg0 * 8;
    const ushort_t* g1 = Bt + (size_t)(n0 + srow1) * K + sseg1 * 8;
    async_copy_b128(g0, &ldsB[0][srow0 * BROW + sseg0 * 8]);
    async_copy_b128(g1, &ldsB[0][srow1 * BROW + sseg1 * 8]);
  }

  for (int kb = 0; kb < ksteps; ++kb) {
    wait_async_done();
    __syncthreads();   // staged slice kb visible to all waves

    if (kb + 1 < ksteps) {  // stage next slice into the other buffer
      const int nb = (kb + 1) & 1;
      const size_t koff = (size_t)(kb + 1) * 64;
      const ushort_t* g0 = Bt + (size_t)(n0 + srow0) * K + koff + sseg0 * 8;
      const ushort_t* g1 = Bt + (size_t)(n0 + srow1) * K + koff + sseg1 * 8;
      async_copy_b128(g0, &ldsB[nb][srow0 * BROW + sseg0 * 8]);
      async_copy_b128(g1, &ldsB[nb][srow1 * BROW + sseg1 * 8]);
    }

    const ushort_t* bbuf = ldsB[kb & 1];
#pragma unroll
    for (int j = 0; j < 2; ++j) {   // two 32-deep WMMA steps per slice
      const int k0 = (kb << 6) + (j << 5);
      // A fragments (ISA 7.12.2 16-bit A 16x32 layout): lane half 0 holds
      // K {0..7, 16..23}, half 1 holds K {8..15, 24..31}.
      ABfrag a0, a1;
      const int ka = k0 + half * 8;
      a0.u[0] = *(const uint4*)(arow0 + ka);
      a0.u[1] = *(const uint4*)(arow0 + ka + 16);
      a1.u[0] = *(const uint4*)(arow1 + ka);
      a1.u[1] = *(const uint4*)(arow1 + ka + 16);
#pragma unroll
      for (int nt = 0; nt < 4; ++nt) {
        // B fragment: lane = column l16 of tile nt, K {0..7,16..23} (+8 for half 1) from LDS
        ABfrag b;
        const ushort_t* brow =
            bbuf + (nt * 16 + l16) * BROW + (j << 5) + half * 8;
        b.u[0] = ((const uint4*)brow)[0];
        b.u[1] = ((const uint4*)(brow + 16))[0];
        acc[0][nt] = __builtin_amdgcn_wmma_f32_16x16x32_bf16(
            false, a0.v, false, b.v, (short)0, acc[0][nt], false, false);
        acc[1][nt] = __builtin_amdgcn_wmma_f32_16x16x32_bf16(
            false, a1.v, false, b.v, (short)0, acc[1][nt], false, false);
      }
    }
    __syncthreads();   // all waves done reading slice kb before overwrite
  }

  // C/D layout: VGPR j -> row m + j + half*8, col n0 + nt*16 + l16
#pragma unroll
  for (int mt = 0; mt < 2; ++mt) {
#pragma unroll
    for (int nt = 0; nt < 4; ++nt) {
      const int col = n0 + nt * 16 + l16;
      const float bv = bias ? bias[col] : 0.0f;
#pragma unroll
      for (int j = 0; j < 8; ++j) {
        const int row = m0 + mt * 16 + j + half * 8;
        C[(size_t)row * N + col] = acc[mt][nt][j] + bv;
      }
    }
  }
}

// ---------------------------------------------------------------------------
// Deformable sampling + attention.
// grid = (B*N), block = (32, H). One wave per (b, n, h); lane owns 2 channels.
// kv layout: row = (b*576 + s)*3 + t, 1536 cols (K: h*64+d, V: 768+h*64+d).
// ---------------------------------------------------------------------------
__global__ __launch_bounds__(384) void deform_attend_kernel(
    const float* __restrict__ kv, const float* __restrict__ qout,
    const float* __restrict__ offout, const float* __restrict__ boff,
    const float* __restrict__ mask, const float* __restrict__ suppress_p,
    float* __restrict__ out) {
  const int lane = threadIdx.x;       // 0..31
  const int h = threadIdx.y;          // 0..11
  const int bn = blockIdx.x;          // 0..2303
  const int b = bn / NNN;
  const int n = bn % NNN;

  const float suppress = suppress_p[0];
  const float mval = mask[bn];
  const float sw = 1.0f - suppress * mval;
  const float moff = suppress * mval * 0.1f;

  const int ri = n / WPD, rj = n % WPD;
  const float refx = (float)rj / (float)(WPD - 1) * 2.0f - 1.0f;
  const float refy = (float)ri / (float)(HPD - 1) * 2.0f - 1.0f;

  const float* qp = qout + (size_t)bn * CCC + h * DHH + lane * 2;
  const float qx = qp[0], qy = qp[1];

  float2 vs[TTT][PPP];
  float lg[TTT][PPP];

#pragma unroll
  for (int p = 0; p < PPP; ++p) {
    const int oc = h * (PPP * 2) + p * 2;
    const float ox = offout[(size_t)bn * 256 + oc + 0] + boff[oc + 0] - moff;
    const float oy = offout[(size_t)bn * 256 + oc + 1] + boff[oc + 1] - moff;
    const float xs = (refx + ox + 1.0f) * 0.5f * (float)(WPD - 1);
    const float ys = (refy + oy + 1.0f) * 0.5f * (float)(HPD - 1);
    const float x0f = floorf(xs), y0f = floorf(ys);
    const int x0 = (int)x0f, y0 = (int)y0f;
    const float wx1 = xs - x0f, wy1 = ys - y0f;

    const int   ixs[4] = { x0, x0 + 1, x0, x0 + 1 };
    const int   iys[4] = { y0, y0, y0 + 1, y0 + 1 };
    const float cws[4] = { (1.0f - wx1) * (1.0f - wy1), wx1 * (1.0f - wy1),
                           (1.0f - wx1) * wy1,          wx1 * wy1 };

#pragma unroll
    for (int t = 0; t < TTT; ++t) {
      float kxa = 0.0f, kya = 0.0f, vxa = 0.0f, vya = 0.0f;
#pragma unroll
      for (int c = 0; c < 4; ++c) {
        const int ix = ixs[c], iy = iys[c];
        if (ix >= 0 && ix < WPD && iy >= 0 && iy < HPD) {
          const int s = iy * WPD + ix;
          const float* base =
              kv + ((size_t)((b * NNN + s) * TTT + t)) * 1536 + h * DHH + lane * 2;
          const float2 kk = *(const float2*)base;
          const float2 vv = *(const float2*)(base + CCC);
          const float w = cws[c];
          kxa += w * kk.x; kya += w * kk.y;
          vxa += w * vv.x; vya += w * vv.y;
        }
      }
      vs[t][p] = make_float2(vxa, vya);
      float pl = qx * kxa + qy * kya;
      // wave32 all-reduce (sum over DH across 32 lanes x 2 ch/lane)
#pragma unroll
      for (int o = 16; o > 0; o >>= 1) pl += __shfl_xor(pl, o, 32);
      lg[t][p] = pl;
    }
  }

  float accx = 0.0f, accy = 0.0f;
#pragma unroll
  for (int t = 0; t < TTT; ++t) {
    float mx = -3.4e38f;
#pragma unroll
    for (int p = 0; p < PPP; ++p) {
      const float l = lg[t][p] * SCALE_F * sw;
      mx = fmaxf(mx, l);
    }
    float e[PPP]; float ssum = 0.0f;
#pragma unroll
    for (int p = 0; p < PPP; ++p) {
      e[p] = __expf(lg[t][p] * SCALE_F * sw - mx);
      ssum += e[p];
    }
    const float inv = 1.0f / ssum;
#pragma unroll
    for (int p = 0; p < PPP; ++p) {
      const float w = e[p] * inv;
      accx += w * vs[t][p].x;
      accy += w * vs[t][p].y;
    }
  }

  float* op = out + (size_t)bn * CCC + h * DHH + lane * 2;
  op[0] = accx * (1.0f / (float)TTT);
  op[1] = accy * (1.0f / (float)TTT);
}

// ---------------------------------------------------------------------------
// Host launcher
// ---------------------------------------------------------------------------
extern "C" void kernel_launch(void* const* d_in, const int* in_sizes, int n_in,
                              void* d_out, int out_size, void* d_ws, size_t ws_size,
                              hipStream_t stream) {
  const float* x        = (const float*)d_in[0];  // (4,576,3,768)
  const float* mask     = (const float*)d_in[1];  // (4,576)
  const float* Wq       = (const float*)d_in[2];  // (768,768)
  const float* Wk       = (const float*)d_in[3];
  const float* Wv       = (const float*)d_in[4];
  const float* Woff     = (const float*)d_in[5];  // (768,216)
  const float* boff     = (const float*)d_in[6];  // (216)
  const float* Wproj    = (const float*)d_in[7];  // (768,768)
  const float* bproj    = (const float*)d_in[8];  // (768)
  const float* suppress = (const float*)d_in[9];  // scalar
  float* out = (float*)d_out;                     // (4,576,768)

  const int M_kv  = BB * NNN * TTT;   // 6912
  const int M_q   = BB * NNN;         // 2304
  const int K     = CCC;              // 768
  const int N_kv  = 2 * CCC;          // 1536
  const int N_off = 256;              // 216 padded to 256

  // workspace layout
  char* ws = (char*)d_ws;
  size_t off = 0;
  auto take = [&](size_t bytes) -> char* {
    char* p = ws + off;
    off = (off + bytes + 255) & ~(size_t)255;
    return p;
  };
  ushort_t* xbf    = (ushort_t*)take((size_t)M_kv * K * 2);
  ushort_t* qbf    = (ushort_t*)take((size_t)M_q * K * 2);
  ushort_t* wkvT   = (ushort_t*)take((size_t)N_kv * K * 2);
  ushort_t* wqT    = (ushort_t*)take((size_t)CCC * K * 2);
  ushort_t* woffT  = (ushort_t*)take((size_t)N_off * K * 2);
  ushort_t* wprojT = (ushort_t*)take((size_t)CCC * K * 2);
  float*    kvout  = (float*)take((size_t)M_kv * N_kv * 4);
  float*    qout   = (float*)take((size_t)M_q * CCC * 4);
  float*    offout = (float*)take((size_t)M_q * N_off * 4);
  float*    attnf  = (float*)take((size_t)M_q * CCC * 4);
  ushort_t* attnbf = (ushort_t*)take((size_t)M_q * CCC * 2);
  (void)ws_size; (void)in_sizes; (void)n_in; (void)out_size;

  const int TPB = 256;
  auto blocks = [&](int n) { return (n + TPB - 1) / TPB; };

  // 1) conversions
  cvt_bf16_kernel<<<blocks(M_kv * K), TPB, 0, stream>>>(x, xbf, M_kv * K);
  extract_query_bf16<<<blocks(M_q * K), TPB, 0, stream>>>(x, qbf, M_q * K);
  transpose_w_bf16<<<blocks(CCC * K), TPB, 0, stream>>>(Wk, wkvT, K, CCC, CCC);
  transpose_w_bf16<<<blocks(CCC * K), TPB, 0, stream>>>(Wv, wkvT + (size_t)CCC * K, K, CCC, CCC);
  transpose_w_bf16<<<blocks(CCC * K), TPB, 0, stream>>>(Wq, wqT, K, CCC, CCC);
  transpose_w_bf16<<<blocks(N_off * K), TPB, 0, stream>>>(Woff, woffT, K, HHH * PPP * 2, N_off);
  transpose_w_bf16<<<blocks(CCC * K), TPB, 0, stream>>>(Wproj, wprojT, K, CCC, CCC);

  // 2) WMMA GEMMs (block tile 256x64)
  gemm_bf16_wmma<<<dim3(N_kv / 64, M_kv / 256), 256, 0, stream>>>(
      xbf, wkvT, kvout, nullptr, M_kv, N_kv, K);
  gemm_bf16_wmma<<<dim3(CCC / 64, M_q / 256), 256, 0, stream>>>(
      qbf, wqT, qout, nullptr, M_q, CCC, K);
  gemm_bf16_wmma<<<dim3(N_off / 64, M_q / 256), 256, 0, stream>>>(
      qbf, woffT, offout, nullptr, M_q, N_off, K);

  // 3) deformable sampling + attention
  deform_attend_kernel<<<dim3(BB * NNN), dim3(32, HHH), 0, stream>>>(
      kvout, qout, offout, boff, mask, suppress, attnf);

  // 4) output projection
  cvt_bf16_kernel<<<blocks(M_q * CCC), TPB, 0, stream>>>(attnf, attnbf, M_q * CCC);
  gemm_bf16_wmma<<<dim3(CCC / 64, M_q / 256), 256, 0, stream>>>(
      attnbf, wprojT, out, bproj, M_q, CCC, K);
}
